// IPA_41747082117334
// MI455X (gfx1250) — matrix operand order repaired and checked by reading, not compiled, not run on hardware
//
#include <hip/hip_runtime.h>
#include <hip/hip_bf16.h>

typedef __attribute__((ext_vector_type(16))) _Float16 v16h;
typedef __attribute__((ext_vector_type(8)))  _Float16 v8h;
typedef __attribute__((ext_vector_type(8)))  float    v8f;

union V16U { v16h v; v8h h[2]; };

#define NPTS 768
#define KPAD 96          // packed K/Q/value feature dim (80/84 used, padded)
#define NEG_MAX (-3.402823466e38f)

// ---------------------------------------------------------------------------
// Kernel 1: per-node prep.
// Builds Kp[h][n][96] (f16) = [s(32) | sqrt2*(Wk1 x_v + coord)(48) | 0..],
//        Qp likewise with Wq1, nq = |vq|^2 per head (nk dropped: softmax is
//        invariant to per-row constants),
//        VmT[h][c][n] (f16) = value matrix transposed:
//            c in [0,32):  vs = x_s @ Wv0 (head slice)
//            c in [32,80): vv_feat = Wv1[:V] applied to x_v
//            c in [80,83): coord, c==83: 1.0, rest 0.
// ---------------------------------------------------------------------------
__global__ __launch_bounds__(128) void ipa_prep(
    const float* __restrict__ xs, const float* __restrict__ xv,
    const float* __restrict__ coord,
    const float* __restrict__ Wk1, const float* __restrict__ Wq1,
    const float* __restrict__ Wv0, const float* __restrict__ Wv1,
    _Float16* __restrict__ Kp, _Float16* __restrict__ Qp,
    _Float16* __restrict__ VmT, float* __restrict__ nq)
{
    const int n = blockIdx.x;
    const int t = threadIdx.x;
    __shared__ float sxv[96], sxs[64], svvf[96], snq[2];

    for (int idx = t; idx < 96; idx += 128) sxv[idx] = xv[n * 96 + idx];
    for (int idx = t; idx < 64; idx += 128) sxs[idx] = xs[n * 64 + idx];
    if (t < 2) snq[t] = 0.f;
    __syncthreads();

    if (t < 96) {
        const int w = t / 3, c = t % 3;
        const float crd = coord[n * 3 + c];
        float kv = 0.f, qv = 0.f, vf = 0.f;
        #pragma unroll
        for (int v = 0; v < 32; ++v) {
            const float x = sxv[v * 3 + c];
            kv += x * Wk1[v * 32 + w];
            qv += x * Wq1[v * 32 + w];
            vf += x * Wv1[v * 32 + w];
        }
        const float vkv = kv + crd, vqv = qv + crd;
        svvf[t] = vf;
        atomicAdd(&snq[w >> 4], vqv * vqv);
        const int hh = w >> 4, d = w & 15;
        const float RT2 = 1.41421356237309515f;
        const size_t base = ((size_t)hh * NPTS + n) * KPAD + 32 + d * 3 + c;
        Kp[base] = (_Float16)(RT2 * vkv);
        Qp[base] = (_Float16)(RT2 * vqv);
    }
    if (t < 64) {
        const int h = t >> 5, cc = t & 31;
        float vsv = 0.f;
        #pragma unroll
        for (int k = 0; k < 64; ++k) vsv += sxs[k] * Wv0[k * 64 + t];
        VmT[((size_t)h * KPAD + cc) * NPTS + n] = (_Float16)vsv;
        const size_t base = ((size_t)h * NPTS + n) * KPAD + cc;
        Kp[base] = (_Float16)sxs[t];
        Qp[base] = (_Float16)sxs[t];
    }
    __syncthreads();

    if (t < 96) {
        const int w = t / 3, c = t % 3;
        const int hh = w >> 4, d = w & 15;
        VmT[((size_t)hh * KPAD + 32 + d * 3 + c) * NPTS + n] = (_Float16)svvf[t];
    }
    if (t < 6) {
        const int h = t / 3, c = t % 3;
        VmT[((size_t)h * KPAD + 80 + c) * NPTS + n] = (_Float16)coord[n * 3 + c];
    }
    if (t < 2) {
        VmT[((size_t)t * KPAD + 83) * NPTS + n] = (_Float16)1.0f;
        nq[t * NPTS + n] = snq[t];
    }
    if (t < 24) {
        const int h = t / 12, c = 84 + (t % 12);
        VmT[((size_t)h * KPAD + c) * NPTS + n] = (_Float16)0.0f;
    }
    if (t < 32) {
        const int h = t >> 4, c = 80 + (t & 15);
        const size_t base = ((size_t)h * NPTS + n) * KPAD + c;
        Kp[base] = (_Float16)0.0f;
        Qp[base] = (_Float16)0.0f;
    }
}

// ---------------------------------------------------------------------------
// Kernel 2: fused attention. One block per (head, 16-row i-tile); 4 waves.
// GEMM1 (WMMA f16): scores strip [16 x 768] = Kp_i . Qp_j^T - nq_j -> LDS f32
// row-max, then e = exp(s - max) stored as f16 in attH (vectorized b128 path),
// GEMM2 (WMMA f16): aggr[16 x 96] = e[16 x 768] . Vm[768 x 96], with the
// 1/rowsum normalization and row mask folded into the epilogue scale.
// A-operand lane layout (16x32 f16): lane L holds row m=L%16, half t maps to
//   k = (t&7) + 8*(2*(t>>3) + (L>>4))  -> two contiguous 8-half runs.
// B-operand lane layout (32x16 f16): lane L holds col n=L%16, k = (t&7) + 8*(2*(t>>3) + (L>>4)).
// ---------------------------------------------------------------------------
__global__ __launch_bounds__(128) void ipa_attn(
    const _Float16* __restrict__ Kp, const _Float16* __restrict__ Qp,
    const _Float16* __restrict__ VmT, const float* __restrict__ nq,
    const int* __restrict__ mask, float* __restrict__ aggr)
{
    const int h  = blockIdx.x / 48;
    const int i0 = (blockIdx.x % 48) * 16;
    const int tid  = threadIdx.x;
    const int wave = tid >> 5;
    const int lane = tid & 31;
    const int lhalf = lane >> 4;   // 0 or 1
    const int llow  = lane & 15;

    __shared__ float sc[16][776];                               // raw scores (f32)
    __shared__ __attribute__((aligned(16))) _Float16 attH[16][784]; // e (f16)
    __shared__ float red[16][8];
    __shared__ float sscale[16];
    __shared__ int   smaskI[16];

    if (tid < 16) smaskI[tid] = (mask[i0 + tid] != 0);

    // Preload A fragments: Kp rows of this i-tile (96 = 3 k-chunks of 32)
    const _Float16* KpRow = Kp + ((size_t)h * NPTS + i0 + llow) * KPAD;
    V16U afrag[3];
    #pragma unroll
    for (int kc = 0; kc < 3; ++kc) {
        const int off = kc * 32 + 8 * lhalf;
        afrag[kc].h[0] = *(const v8h*)(KpRow + off);
        afrag[kc].h[1] = *(const v8h*)(KpRow + off + 16);
    }

    // ---- GEMM1: scores ----
    for (int jt = wave; jt < 48; jt += 4) {
        const int j0 = jt * 16;
        const _Float16* QpRow = Qp + ((size_t)h * NPTS + j0 + llow) * KPAD;
        v8f acc = {};
        #pragma unroll
        for (int kc = 0; kc < 3; ++kc) {
            V16U b;
            const int off = kc * 32 + 8 * lhalf;
            b.h[0] = *(const v8h*)(QpRow + off);
            b.h[1] = *(const v8h*)(QpRow + off + 16);
            acc = __builtin_amdgcn_wmma_f32_16x16x32_f16(
                false, afrag[kc].v, false, b.v, (short)0, acc, false, false);
        }
        const int jj = j0 + llow;
        const int mj = (mask[jj] != 0);
        const float nqv = nq[h * NPTS + jj];
        #pragma unroll
        for (int r = 0; r < 8; ++r) {
            const int m = r + 8 * lhalf;
            sc[m][jj] = mj ? (acc[r] - nqv) : NEG_MAX;
        }
    }
    __syncthreads();

    // ---- row max, then e = exp(s - max) -> f16 attH ----
    const int row = tid >> 3, sub = tid & 7;
    const int jbeg = sub * 96;
    float lm = NEG_MAX;
    for (int j = jbeg; j < jbeg + 96; ++j) lm = fmaxf(lm, sc[row][j]);
    red[row][sub] = lm;
    __syncthreads();
    float rm = red[row][0];
    #pragma unroll
    for (int k = 1; k < 8; ++k) rm = fmaxf(rm, red[row][k]);

    float ls = 0.f;
    for (int jb = jbeg; jb < jbeg + 96; jb += 8) {
        v8h ev;
        #pragma unroll
        for (int t = 0; t < 8; ++t) {
            const float sv = sc[row][jb + t];
            const float e = (sv == NEG_MAX) ? 0.f : __expf(sv - rm);
            ls += e;
            ev[t] = (_Float16)e;
        }
        *(v8h*)&attH[row][jb] = ev;
    }
    __syncthreads();
    red[row][sub] = ls;
    __syncthreads();
    if (sub == 0) {
        float rs = 0.f;
        #pragma unroll
        for (int k = 0; k < 8; ++k) rs += red[row][k];
        sscale[row] = (smaskI[row] && rs > 0.f) ? 1.f / rs : 0.f;
    }
    __syncthreads();

    // ---- GEMM2: aggr = (e . Vm) * sscale ----
    for (int nt = wave; nt < 6; nt += 4) {
        const int c = nt * 16 + llow;
        const _Float16* Brow = VmT + ((size_t)h * KPAD + c) * NPTS;
        v8f acc = {};
        for (int kc = 0; kc < 24; ++kc) {
            const int j0 = kc * 32;
            V16U a2;
            a2.h[0] = *(const v8h*)&attH[llow][j0 + 8 * lhalf];
            a2.h[1] = *(const v8h*)&attH[llow][j0 + 16 + 8 * lhalf];
            V16U b;
            const int off = j0 + 8 * lhalf;
            b.h[0] = *(const v8h*)(Brow + off);
            b.h[1] = *(const v8h*)(Brow + off + 16);
            acc = __builtin_amdgcn_wmma_f32_16x16x32_f16(
                false, a2.v, false, b.v, (short)0, acc, false, false);
        }
        #pragma unroll
        for (int r = 0; r < 8; ++r) {
            const int m = r + 8 * lhalf;
            aggr[((size_t)h * NPTS + i0 + m) * KPAD + c] = acc[r] * sscale[m];
        }
    }
}

// ---------------------------------------------------------------------------
// Kernel 3: finalize. aggr -> rank-1 coord fix -> Wo projections -> residual
// -> scalar LN + vector RMS norm -> out [768,160].
// ---------------------------------------------------------------------------
__global__ __launch_bounds__(256) void ipa_finalize(
    const float* __restrict__ xs, const float* __restrict__ xv,
    const float* __restrict__ coord, const float* __restrict__ Wv1,
    const float* __restrict__ Wo0, const float* __restrict__ Wo1,
    const float* __restrict__ aggr, float* __restrict__ out)
{
    const int i = blockIdx.x * 256 + threadIdx.x;
    if (i >= NPTS) return;
    const float* ag[2] = { aggr + ((size_t)0 * NPTS + i) * KPAD,
                           aggr + ((size_t)1 * NPTS + i) * KPAD };

    // --- scalars ---
    {
        float as_[64];
        #pragma unroll
        for (int c = 0; c < 32; ++c) { as_[c] = ag[0][c]; as_[32 + c] = ag[1][c]; }
        float ys[64], sum = 0.f, sq = 0.f;
        for (int c = 0; c < 64; ++c) {
            float acc = xs[i * 64 + c];
            #pragma unroll
            for (int k = 0; k < 64; ++k) acc += as_[k] * Wo0[k * 64 + c];
            ys[c] = acc; sum += acc; sq += acc * acc;
        }
        const float mean = sum * (1.f / 64.f);
        const float var  = sq * (1.f / 64.f) - mean * mean;
        const float s    = rsqrtf(var + 1e-6f);
        for (int c = 0; c < 64; ++c) out[(size_t)i * 160 + c] = (ys[c] - mean) * s;
    }
    // --- vectors ---
    {
        float av[96];
        #pragma unroll
        for (int h = 0; h < 2; ++h) {
            const float rsm = ag[h][83];
            float cs[3];
            #pragma unroll
            for (int c = 0; c < 3; ++c) cs[c] = ag[h][80 + c] - coord[i * 3 + c] * rsm;
            #pragma unroll
            for (int d = 0; d < 16; ++d) {
                const float wv = Wv1[32 * 32 + h * 16 + d];
                #pragma unroll
                for (int c = 0; c < 3; ++c)
                    av[(h * 16 + d) * 3 + c] = ag[h][32 + d * 3 + c] + wv * cs[c];
            }
        }
        float yv[96], s2 = 0.f;
        for (int w = 0; w < 32; ++w) {
            #pragma unroll
            for (int c = 0; c < 3; ++c) {
                float acc = xv[i * 96 + w * 3 + c];
                #pragma unroll
                for (int v = 0; v < 32; ++v) acc += av[v * 3 + c] * Wo1[v * 32 + w];
                yv[w * 3 + c] = acc; s2 += acc * acc;
            }
        }
        const float rinv = rsqrtf(s2 * (1.f / 32.f) + 1e-6f);
        for (int k = 0; k < 96; ++k) out[(size_t)i * 160 + 64 + k] = yv[k] * rinv;
    }
}

// ---------------------------------------------------------------------------
extern "C" void kernel_launch(void* const* d_in, const int* in_sizes, int n_in,
                              void* d_out, int out_size, void* d_ws, size_t ws_size,
                              hipStream_t stream) {
    const float* xs    = (const float*)d_in[0];
    const float* xv    = (const float*)d_in[1];
    const float* coord = (const float*)d_in[2];
    const int*   mask  = (const int*)d_in[3];
    const float* Wk1   = (const float*)d_in[4];
    const float* Wq1   = (const float*)d_in[5];
    const float* Wv0   = (const float*)d_in[6];
    const float* Wv1   = (const float*)d_in[7];
    const float* Wo0   = (const float*)d_in[8];
    const float* Wo1   = (const float*)d_in[9];
    float* out = (float*)d_out;

    char* ws = (char*)d_ws;
    const size_t szKp = (size_t)2 * NPTS * KPAD * sizeof(_Float16); // 294912
    _Float16* Kp   = (_Float16*)(ws);
    _Float16* Qp   = (_Float16*)(ws + szKp);
    _Float16* VmT  = (_Float16*)(ws + 2 * szKp);
    float*    nq   = (float*)(ws + 3 * szKp);
    float*    aggr = (float*)(ws + 3 * szKp + 2 * NPTS * sizeof(float));

    ipa_prep<<<NPTS, 128, 0, stream>>>(xs, xv, coord, Wk1, Wq1, Wv0, Wv1,
                                       Kp, Qp, VmT, nq);
    ipa_attn<<<96, 128, 0, stream>>>(Kp, Qp, VmT, nq, mask, aggr);
    ipa_finalize<<<3, 256, 0, stream>>>(xs, xv, coord, Wv1, Wo0, Wo1, aggr, out);
}
